// init_5952824672725
// MI455X (gfx1250) — matrix operand order repaired and checked by reading, not compiled, not run on hardware
//
#include <hip/hip_runtime.h>
#include <hip/hip_bf16.h>
#include <math.h>

// Problem constants (match reference)
#define N_NODES   100000
#define N_EDGES   800000
#define NUM_RAD   6
#define HID       128
#define NODE_FEAT 48
#define EDGE_TILES_PER_WG 10   // 5000 WGs * 10 tiles * 16 edges = 800000

typedef _Float16 v16h __attribute__((ext_vector_type(16)));
typedef float    v8f  __attribute__((ext_vector_type(8)));
typedef _Float16 h2v  __attribute__((ext_vector_type(2)));

__device__ __forceinline__ float selu_f(float x) {
    const float scale = 1.0507009873554805f;
    const float alpha = 1.6732632423543772f;
    return x > 0.0f ? scale * x : scale * alpha * (__expf(x) - 1.0f);
}

// A-matrix (16x32 f16) register build per ISA 7.12.2:
// lane(0-15) holds row M=lane; VGPR v<4 : K = 2v + 8*half; v>=4 : K = 16 + 2(v-4) + 8*half
__device__ __forceinline__ v16h wmma_load_a(const _Float16* src, int l16, int half,
                                            int kofs, int stride) {
    v16h a;
#pragma unroll
    for (int v = 0; v < 8; ++v) {
        int kb = (v < 4 ? 2 * v : 16 + 2 * (v - 4)) + 8 * half + kofs;
        h2v p = *(const h2v*)(src + l16 * stride + kb);
        a[2 * v]     = p.x;
        a[2 * v + 1] = p.y;
    }
    return a;
}

// B-matrix (32x16 f16): lane holds column N; same K order as A:
// VGPR v<4 : K = 2v + 8*half; v>=4 : K = 16 + 2(v-4) + 8*half. Weights stored transposed as wT[n][k], k-contiguous.
__device__ __forceinline__ v16h wmma_load_b(const _Float16* wT, int n, int half, int kofs) {
    v16h b;
#pragma unroll
    for (int v = 0; v < 8; ++v) {
        int k = kofs + (v < 4 ? 2 * v : 16 + 2 * (v - 4)) + 8 * half;
        h2v p = *(const h2v*)(wT + n * HID + k);
        b[2 * v]     = p.x;
        b[2 * v + 1] = p.y;
    }
    return b;
}

__device__ __forceinline__ v8f wmma_f16(v16h a, v16h b, v8f c) {
    return __builtin_amdgcn_wmma_f32_16x16x32_f16(false, a, false, b, (short)0, c,
                                                  false, false);
}

// ---------------------------------------------------------------------------
// Kernel 0: convert W_lin blocks to f16, transposed [n][k] (B-matrix friendly)
// ---------------------------------------------------------------------------
__global__ __launch_bounds__(256) void prep_weights_kernel(
    const float* __restrict__ W_lin,
    _Float16* __restrict__ WaT, _Float16* __restrict__ WbT, _Float16* __restrict__ WcT) {
    int idx = blockIdx.x * 256 + threadIdx.x;
    if (idx < HID * HID) {
        int n = idx >> 7;
        int k = idx & 127;
        const float* row = W_lin + (size_t)n * (3 * HID);
        WaT[idx] = (_Float16)row[k];
        WbT[idx] = (_Float16)row[HID + k];
        WcT[idx] = (_Float16)row[2 * HID + k];
    }
}

// ---------------------------------------------------------------------------
// Kernel 1: per-node  xh = x@Wn.T + bn (f32 VALU, K=48) ;  ya = xh@Wa.T,
//           yb = xh@Wb.T via WMMA f16. One 16-node tile per workgroup.
//           ya/yb are written with default (temporal) policy: they are the
//           L2-resident gather tables for the edge kernel.
// ---------------------------------------------------------------------------
__global__ __launch_bounds__(256) void node_kernel(
    const float* __restrict__ x, const float* __restrict__ W_node,
    const float* __restrict__ b_node,
    const _Float16* __restrict__ WaT, const _Float16* __restrict__ WbT,
    float* __restrict__ ya, float* __restrict__ yb) {
    const int tid  = threadIdx.x;
    const int wave = tid >> 5;
    const int lane = tid & 31;
    const int half = lane >> 4;
    const int l16  = lane & 15;
    const int n0   = blockIdx.x * 16;

    __shared__ __align__(16) float    sX[16 * NODE_FEAT];
    __shared__ __align__(16) _Float16 sXH[16 * HID];

    // stage node-feature tile (x is read exactly once -> non-temporal)
    for (int idx = tid; idx < 16 * NODE_FEAT; idx += 256) {
        int m = idx / NODE_FEAT, c = idx % NODE_FEAT;
        sX[idx] = __builtin_nontemporal_load(x + (size_t)(n0 + m) * NODE_FEAT + c);
    }
    __syncthreads();

    // xh = x @ W_node.T + b_node  (exact f32), write f16 into LDS (A layout: row-major)
    {
        int m  = tid >> 4;
        int h0 = (tid & 15) * 8;
        float xv[NODE_FEAT];
#pragma unroll
        for (int c = 0; c < NODE_FEAT; ++c) xv[c] = sX[m * NODE_FEAT + c];
#pragma unroll
        for (int hh = 0; hh < 8; ++hh) {
            int h = h0 + hh;
            const float* w = W_node + (size_t)h * NODE_FEAT;
            float acc = b_node[h];
#pragma unroll
            for (int c = 0; c < NODE_FEAT; ++c) acc = fmaf(xv[c], w[c], acc);
            sXH[m * HID + h] = (_Float16)acc;
        }
    }
    __syncthreads();

    // each wave owns a 16-column N-tile for BOTH ya and yb
    const int n = wave * 16 + l16;
    v8f accA = {};
    v8f accB = {};
#pragma unroll
    for (int ks = 0; ks < 4; ++ks) {
        v16h A  = wmma_load_a(sXH, l16, half, ks * 32, HID);
        v16h Ba = wmma_load_b(WaT, n, half, ks * 32);
        v16h Bb = wmma_load_b(WbT, n, half, ks * 32);
        accA = wmma_f16(A, Ba, accA);
        accB = wmma_f16(A, Bb, accB);
    }
#pragma unroll
    for (int r = 0; r < 8; ++r) {
        int m = r + 8 * half;
        size_t o = (size_t)(n0 + m) * HID + n;
        ya[o] = accA[r];
        yb[o] = accB[r];
    }
}

// ---------------------------------------------------------------------------
// Kernel 2: per-edge.  rbf0 = selu(rbf@W0.T+b0) (f32 VALU, K=6) -> LDS f16;
//           gate = rbf@W1.T -> LDS f32;   t = rbf0@Wc.T via WMMA;
//           e1 = selu(t + ya[i] + yb[j] + b_lin);  e2 = gate * e1.
//           Streaming I/O (rbf, i, j, e1, e2) is non-temporal so the 819 MB
//           output stream does not evict the 102 MB ya/yb tables from L2.
// ---------------------------------------------------------------------------
__global__ __launch_bounds__(256) void edge_kernel(
    const float* __restrict__ rbf, const float* __restrict__ W_rbf0,
    const float* __restrict__ b_rbf0, const float* __restrict__ W_rbf1,
    const float* __restrict__ b_lin,
    const int* __restrict__ ei, const int* __restrict__ ej,
    const float* __restrict__ ya, const float* __restrict__ yb,
    const _Float16* __restrict__ WcT,
    float* __restrict__ e1_out, float* __restrict__ e2_out) {
    const int tid  = threadIdx.x;
    const int wave = tid >> 5;
    const int lane = tid & 31;
    const int half = lane >> 4;
    const int l16  = lane & 15;

    __shared__ __align__(16) _Float16 sA[16 * HID];  // rbf0 (f16), 16 edges x 128
    __shared__ __align__(16) float    sG[16 * HID];  // gate  (f32)
    __shared__ __align__(16) float    sR[16 * NUM_RAD];
    __shared__ int sIJ[32];

    // hoist per-wave invariants: B registers (Wc columns) + bias
    const int n = wave * 16 + l16;
    const v16h B0 = wmma_load_b(WcT, n, half, 0);
    const v16h B1 = wmma_load_b(WcT, n, half, 32);
    const v16h B2 = wmma_load_b(WcT, n, half, 64);
    const v16h B3 = wmma_load_b(WcT, n, half, 96);
    const float bl = b_lin[n];

    for (int t = 0; t < EDGE_TILES_PER_WG; ++t) {
        const int tile = blockIdx.x * EDGE_TILES_PER_WG + t;
        const int e0   = tile * 16;
        if (e0 >= N_EDGES) break;

        __syncthreads();  // previous epilogue done (sG/sIJ reads complete)
        if (tid < 16 * NUM_RAD)
            sR[tid] = __builtin_nontemporal_load(rbf + (size_t)e0 * NUM_RAD + tid);
        if (tid < 16) {
            sIJ[tid]      = __builtin_nontemporal_load(ei + e0 + tid);
            sIJ[16 + tid] = __builtin_nontemporal_load(ej + e0 + tid);
        }
        __syncthreads();

        // stage 1: tiny K=6 projections in exact f32 (256 thr x 8 outputs)
        {
            int m  = tid >> 4;
            int h0 = (tid & 15) * 8;
            float rv[NUM_RAD];
#pragma unroll
            for (int c = 0; c < NUM_RAD; ++c) rv[c] = sR[m * NUM_RAD + c];
#pragma unroll
            for (int hh = 0; hh < 8; ++hh) {
                int h = h0 + hh;
                const float* w0 = W_rbf0 + (size_t)h * NUM_RAD;
                const float* w1 = W_rbf1 + (size_t)h * NUM_RAD;
                float a0 = b_rbf0[h];
                float a1 = 0.0f;
#pragma unroll
                for (int c = 0; c < NUM_RAD; ++c) {
                    a0 = fmaf(rv[c], w0[c], a0);
                    a1 = fmaf(rv[c], w1[c], a1);
                }
                sA[m * HID + h] = (_Float16)selu_f(a0);
                sG[m * HID + h] = a1;
            }
        }
        __syncthreads();

        // stage 2: 16x128 @ 128x128 via 4 chained WMMAs per wave (uniform EXEC)
        v8f acc = {};
        v16h A;
        A = wmma_load_a(sA, l16, half, 0, HID);   acc = wmma_f16(A, B0, acc);
        A = wmma_load_a(sA, l16, half, 32, HID);  acc = wmma_f16(A, B1, acc);
        A = wmma_load_a(sA, l16, half, 64, HID);  acc = wmma_f16(A, B2, acc);
        A = wmma_load_a(sA, l16, half, 96, HID);  acc = wmma_f16(A, B3, acc);

        // epilogue: gather ya[i], yb[j] (temporal, L2-resident tables),
        // selu, gate, then stream the outputs out non-temporally.
#pragma unroll
        for (int r = 0; r < 8; ++r) {
            int m  = r + 8 * half;
            int ni = sIJ[m];
            int nj = sIJ[16 + m];
            float pre = acc[r] + ya[(size_t)ni * HID + n] + yb[(size_t)nj * HID + n] + bl;
            float v1  = selu_f(pre);
            float v2  = sG[m * HID + n] * v1;
            size_t o  = (size_t)(e0 + m) * HID + n;
            __builtin_nontemporal_store(v1, e1_out + o);
            __builtin_nontemporal_store(v2, e2_out + o);
        }
    }
}

extern "C" void kernel_launch(void* const* d_in, const int* in_sizes, int n_in,
                              void* d_out, int out_size, void* d_ws, size_t ws_size,
                              hipStream_t stream) {
    const float* x      = (const float*)d_in[0];
    const float* rbf    = (const float*)d_in[1];
    const float* W_node = (const float*)d_in[2];
    const float* b_node = (const float*)d_in[3];
    const float* W_rbf0 = (const float*)d_in[4];
    const float* b_rbf0 = (const float*)d_in[5];
    const float* W_lin  = (const float*)d_in[6];
    const float* b_lin  = (const float*)d_in[7];
    const float* W_rbf1 = (const float*)d_in[8];
    const int*   ei     = (const int*)d_in[9];
    const int*   ej     = (const int*)d_in[10];

    float* e1 = (float*)d_out;
    float* e2 = e1 + (size_t)N_EDGES * HID;

    // workspace layout: ya, yb (f32 node tables, L2-resident) + f16 weight blocks
    float*    ya  = (float*)d_ws;                         // N_NODES*128 f32
    float*    yb  = ya + (size_t)N_NODES * HID;           // N_NODES*128 f32
    _Float16* WaT = (_Float16*)(yb + (size_t)N_NODES * HID);
    _Float16* WbT = WaT + HID * HID;
    _Float16* WcT = WbT + HID * HID;
    // total ws use: 2*51.2MB + 96KB  (~102.5 MB)

    prep_weights_kernel<<<(HID * HID + 255) / 256, 256, 0, stream>>>(W_lin, WaT, WbT, WcT);
    node_kernel<<<N_NODES / 16, 256, 0, stream>>>(x, W_node, b_node, WaT, WbT, ya, yb);
    edge_kernel<<<N_EDGES / (16 * EDGE_TILES_PER_WG), 256, 0, stream>>>(
        rbf, W_rbf0, b_rbf0, W_rbf1, b_lin, ei, ej, ya, yb, WcT, e1, e2);
}
